// MultiScaleConvNet_29102698398259
// MI455X (gfx1250) — matrix operand order repaired and checked by reading, not compiled, not run on hardware
//
#include <hip/hip_runtime.h>
#include <math.h>

// ---------------------------------------------------------------------------
// MultiScaleConvNet for MI455X (gfx1250, wave32, WMMA).
//
// CGConv's per-edge GEMM z@W with z=[x_dst,x_src,e] factors into per-NODE
// GEMMs P_d = x@W[0:100], P_s = x@W[100:200] plus a per-edge 3/4-term FMA.
// Dense node GEMMs (K<=128) run on v_wmma_f32_16x16x32_f16 reading f16
// fragments DIRECTLY from global memory: activations/weights are pre-packed
// into zero-padded f16 buffers whose layout matches the WMMA fragment layout
// exactly (aligned b128 loads, no LDS, no bounds checks in the K loop).
// Each wave owns a 32x32 patch -> 4 wmma per K-step.
// ---------------------------------------------------------------------------

#define NND  20000   // nodes
#define MPAD 20032   // nodes padded to multiple of 64 (313 * 64)
#define NE   320000  // edges per scale
#define NB   64      // graphs
#define FIN  92
#define D1   100
#define D2   128
#define KP   128     // padded K pitch (f16 elements)

typedef __attribute__((ext_vector_type(16))) _Float16 v16h;
typedef __attribute__((ext_vector_type(8)))  _Float16 v8h;
typedef __attribute__((ext_vector_type(8)))  float    v8f;
typedef __attribute__((ext_vector_type(4)))  float    v4f;

// ---------------- pack kernels: f32 -> zero-padded f16 ----------------------
// Activations: out[MPAD x KP], row-major, zero outside [M x K].
__global__ void pack_a_kernel(const float* __restrict__ A, int lda, int K,
                              _Float16* __restrict__ out)
{
    int i = blockIdx.x * 256 + threadIdx.x;      // MPAD*KP threads
    int r = i >> 7, c = i & (KP - 1);
    float v = (r < NND && c < K) ? A[(size_t)r * lda + c] : 0.f;
    out[i] = (_Float16)v;
}

// Weights: W is [K x N] row-major f32; out is n-major [KP x KP] f16:
// out[n*KP + k] = W[k*ldw + n], zero-padded.
__global__ void pack_w_kernel(const float* __restrict__ W, int ldw, int K, int N,
                              _Float16* __restrict__ out)
{
    int i = blockIdx.x * 256 + threadIdx.x;      // KP*KP = 16384 threads
    int n = i >> 7, k = i & (KP - 1);
    float v = (n < N && k < K) ? W[(size_t)k * ldw + n] : 0.f;
    out[i] = (_Float16)v;
}

// ---------------- WMMA GEMM, fragments direct from global -------------------
// A fragment (16x32 f16): lane half hf covers K = {8hf..8hf+7, 16+8hf..+7}.
// B fragment (32x16 f16) from n-major W: lane half hf covers K = {8hf..8hf+7, 16+8hf..+7}.
__device__ __forceinline__ v16h load_frag_a(const _Float16* p, int hf) {
    v8h lo = *(const v8h*)(p + 8 * hf);
    v8h hi = *(const v8h*)(p + 16 + 8 * hf);
    v16h r;
#pragma unroll
    for (int e = 0; e < 8; ++e) { r[e] = lo[e]; r[8 + e] = hi[e]; }
    return r;
}
__device__ __forceinline__ v16h load_frag_b(const _Float16* p, int hf) {
    v8h lo = *(const v8h*)(p + 8 * hf);
    v8h hi = *(const v8h*)(p + 16 + 8 * hf);
    v16h r;
#pragma unroll
    for (int e = 0; e < 8; ++e) { r[e] = lo[e]; r[8 + e] = hi[e]; }
    return r;
}

__device__ __forceinline__ void store_tile(v8f acc, int r0, int c0, int l15, int hf,
                                           const float* bias, float* C, int ldc,
                                           int M, int Ncols, int act) {
    int col = c0 + l15;
    if (col < Ncols) {
        float bv = bias ? bias[col] : 0.f;
#pragma unroll
        for (int v = 0; v < 8; ++v) {
            int row = r0 + v + 8 * hf;
            if (row < M) {
                float r = acc[v] + bv;
                if (act == 1) r = fmaxf(r, 0.f);
                C[(size_t)row * ldc + col] = r;
            }
        }
    }
}

// Block: 256 thr = 8 wave32 as 2(m) x 4(n); block tile 64 x 128.
// Each wave: 32x32 patch = 4 accumulators, 4 wmma per K-step.
__global__ __launch_bounds__(256) void gemm_f16_wmma(
    const _Float16* __restrict__ A16,    // [MPAD x KP] zero-padded
    const _Float16* __restrict__ Wt16,   // [KP x KP] n-major zero-padded
    const float* __restrict__ bias,
    float* __restrict__ C, int ldc,
    int M, int Ncols, int kSteps, int act)
{
    const int lane = threadIdx.x & 31;
    const int wave = threadIdx.x >> 5;
    const int hf   = lane >> 4;
    const int l15  = lane & 15;
    const int mbase = blockIdx.x * 64 + (wave & 1) * 32;
    const int nbase = (wave >> 1) * 32;

    const _Float16* a0row = A16 + (size_t)(mbase + l15) * KP;
    const _Float16* a1row = A16 + (size_t)(mbase + 16 + l15) * KP;
    const _Float16* b0row = Wt16 + (size_t)(nbase + l15) * KP;
    const _Float16* b1row = Wt16 + (size_t)(nbase + 16 + l15) * KP;

    v8f acc00 = {}, acc01 = {}, acc10 = {}, acc11 = {};
    for (int kk = 0; kk < kSteps; ++kk) {
        const int k0 = kk * 32;
        v16h a0 = load_frag_a(a0row + k0, hf);
        v16h a1 = load_frag_a(a1row + k0, hf);
        v16h b0 = load_frag_b(b0row + k0, hf);
        v16h b1 = load_frag_b(b1row + k0, hf);
        acc00 = __builtin_amdgcn_wmma_f32_16x16x32_f16(false, a0, false, b0, (short)0, acc00, false, false);
        acc01 = __builtin_amdgcn_wmma_f32_16x16x32_f16(false, a0, false, b1, (short)0, acc01, false, false);
        acc10 = __builtin_amdgcn_wmma_f32_16x16x32_f16(false, a1, false, b0, (short)0, acc10, false, false);
        acc11 = __builtin_amdgcn_wmma_f32_16x16x32_f16(false, a1, false, b1, (short)0, acc11, false, false);
    }
    store_tile(acc00, mbase,      nbase,      l15, hf, bias, C, ldc, M, Ncols, act);
    store_tile(acc01, mbase,      nbase + 16, l15, hf, bias, C, ldc, M, Ncols, act);
    store_tile(acc10, mbase + 16, nbase,      l15, hf, bias, C, ldc, M, Ncols, act);
    store_tile(acc11, mbase + 16, nbase + 16, l15, hf, bias, C, ldc, M, Ncols, act);
}

// ----------------- CGConv edge pass (wave-per-edge, factored) ---------------
// 25 lanes x float4 cover the 100 channels; P rows are 400B => 16B aligned.
__global__ __launch_bounds__(256) void cgconv_edge_kernel(
    const int* __restrict__ src, const int* __restrict__ dst,
    const float* __restrict__ ea, int edim,
    const float* __restrict__ Pfd, const float* __restrict__ Pfs,
    const float* __restrict__ Psd, const float* __restrict__ Pss,
    const float* __restrict__ Wfe, const float* __restrict__ bf,
    const float* __restrict__ Wse, const float* __restrict__ bs,
    float* __restrict__ agg, float* __restrict__ deg, int E)
{
    int e = blockIdx.x * 8 + (threadIdx.x >> 5);
    if (e >= E) return;
    int lane = threadIdx.x & 31;
    int d = dst[e], s = src[e];
    float a0 = ea[(size_t)e * edim + 0];
    float a1 = ea[(size_t)e * edim + 1];
    float a2 = ea[(size_t)e * edim + 2];
    float a3 = (edim == 4) ? ea[(size_t)e * edim + 3] : 0.f;
    if (lane == 0) atomicAdd(&deg[d], 1.f);
    if (lane >= 25) return;
    int c = lane * 4;

    v4f mf = *(const v4f*)(Pfd + (size_t)d * D1 + c);
    v4f t0 = *(const v4f*)(Pfs + (size_t)s * D1 + c);
    v4f ms = *(const v4f*)(Psd + (size_t)d * D1 + c);
    v4f t1 = *(const v4f*)(Pss + (size_t)s * D1 + c);
    v4f wf0 = *(const v4f*)(Wfe + c),       ws0 = *(const v4f*)(Wse + c);
    v4f wf1 = *(const v4f*)(Wfe + D1 + c),  ws1 = *(const v4f*)(Wse + D1 + c);
    v4f wf2 = *(const v4f*)(Wfe + 2*D1 + c),ws2 = *(const v4f*)(Wse + 2*D1 + c);
    v4f bfv = *(const v4f*)(bf + c),        bsv = *(const v4f*)(bs + c);
    v4f wf3 = {}, ws3 = {};
    if (edim == 4) { wf3 = *(const v4f*)(Wfe + 3*D1 + c); ws3 = *(const v4f*)(Wse + 3*D1 + c); }

#pragma unroll
    for (int j = 0; j < 4; ++j) {
        float vf = mf[j] + t0[j] + a0*wf0[j] + a1*wf1[j] + a2*wf2[j] + a3*wf3[j] + bfv[j];
        float vs = ms[j] + t1[j] + a0*ws0[j] + a1*ws1[j] + a2*ws2[j] + a3*ws3[j] + bsv[j];
        float sg = 1.f / (1.f + __expf(-vf));
        float sp = (vs > 20.f) ? vs : log1pf(__expf(vs));
        atomicAdd(&agg[(size_t)d * D1 + c + j], sg * sp);
    }
}

// xout = xin + agg / max(deg,1)
__global__ void combine_kernel(const float* __restrict__ xin,
                               const float* __restrict__ agg,
                               const float* __restrict__ deg,
                               float* __restrict__ xout, int n)
{
    int i = blockIdx.x * 256 + threadIdx.x;
    if (i >= n * D1) return;
    float dg = deg[i / D1];
    dg = dg < 1.f ? 1.f : dg;
    xout[i] = xin[i] + agg[i] / dg;
}

// ---- BatchNorm (training-mode, population var), hierarchical stats ----
// Pass 1: each block reduces ROWS_PER_BLK rows (coalesced row-major reads)
// into LDS channel sums via ds_add_f32, then one global atomic per channel.
#define ROWS_PER_BLK 100
__global__ __launch_bounds__(256) void bn_stats_kernel(const float* __restrict__ x,
                                                       float* __restrict__ ssum,
                                                       float* __restrict__ ssq, int n)
{
    __shared__ float lsum[D1], lsq[D1];
    if (threadIdx.x < D1) { lsum[threadIdx.x] = 0.f; lsq[threadIdx.x] = 0.f; }
    __syncthreads();
    int r0 = blockIdx.x * ROWS_PER_BLK;
    int cnt = (n - r0 < ROWS_PER_BLK) ? (n - r0) : ROWS_PER_BLK;
    for (int i = threadIdx.x; i < cnt * D1; i += 256) {
        float v = x[(size_t)r0 * D1 + i];
        int c = (r0 * D1 + i) % D1;
        atomicAdd(&lsum[c], v);
        atomicAdd(&lsq[c], v * v);
    }
    __syncthreads();
    if (threadIdx.x < D1) {
        atomicAdd(&ssum[threadIdx.x], lsum[threadIdx.x]);
        atomicAdd(&ssq[threadIdx.x],  lsq[threadIdx.x]);
    }
}
// Pass 2: normalize elementwise.
__global__ void bn_apply_kernel(float* __restrict__ x,
                                const float* __restrict__ ssum, const float* __restrict__ ssq,
                                const float* __restrict__ gamma, const float* __restrict__ beta,
                                int n)
{
    int i = blockIdx.x * 256 + threadIdx.x;
    if (i >= n * D1) return;
    int c = i % D1;
    float mu  = ssum[c] / n;
    float var = ssq[c] / n - mu * mu;
    float inv = rsqrtf(var + 1e-5f);
    x[i] = gamma[c] * (x[i] - mu) * inv + beta[c];
}

// Attention scorer node mat-vecs: A1 = x@Wa[0:100], A2 = x@Wa[100:200]
__global__ void att_node_kernel(const float* __restrict__ x,
                                const float* __restrict__ Wa,
                                float* __restrict__ A1, float* __restrict__ A2, int n)
{
    int r = blockIdx.x * 256 + threadIdx.x;
    if (r >= n) return;
    float s1 = 0.f, s2 = 0.f;
    for (int c = 0; c < D1; c += 4) {
        v4f v = *(const v4f*)(x + (size_t)r * D1 + c);
        v4f w1 = *(const v4f*)(Wa + c);
        v4f w2 = *(const v4f*)(Wa + D1 + c);
#pragma unroll
        for (int j = 0; j < 4; ++j) { s1 += v[j] * w1[j]; s2 += v[j] * w2[j]; }
    }
    A1[r] = s1; A2[r] = s2;
}

// Per-edge sigmoid score; writes [ea0,ea1,ea2,score] to combined attr table.
__global__ void edge_score_kernel(const int* __restrict__ src, const int* __restrict__ dst,
                                  const float* __restrict__ ea,
                                  const float* __restrict__ A1, const float* __restrict__ A2,
                                  const float* __restrict__ Wa, const float* __restrict__ ab,
                                  float* __restrict__ ea_out, int E)
{
    int e = blockIdx.x * 256 + threadIdx.x;
    if (e >= E) return;
    float a0 = ea[(size_t)e * 3 + 0], a1 = ea[(size_t)e * 3 + 1], a2 = ea[(size_t)e * 3 + 2];
    float sc = A1[dst[e]] + A2[src[e]]
             + a0 * Wa[2 * D1 + 0] + a1 * Wa[2 * D1 + 1] + a2 * Wa[2 * D1 + 2] + ab[0];
    sc = 1.f / (1.f + __expf(-sc));
    v4f o = {a0, a1, a2, sc};
    *(v4f*)(ea_out + (size_t)e * 4) = o;
}

// Concatenate 3 edge index sets into [2, 3E] (src row, then dst row).
__global__ void concat_edges(const int* __restrict__ e1, const int* __restrict__ e2,
                             const int* __restrict__ e3, int* __restrict__ out)
{
    int i = blockIdx.x * 256 + threadIdx.x;
    if (i >= 3 * NE) return;
    const int* e = (i < NE) ? e1 : (i < 2 * NE ? e2 : e3);
    int off = (i < NE) ? i : (i < 2 * NE ? i - NE : i - 2 * NE);
    out[i]          = e[off];
    out[3 * NE + i] = e[NE + off];
}

// global_mean_pool accumulation
__global__ void pool_kernel(const float* __restrict__ x, const int* __restrict__ batch,
                            float* __restrict__ pooled, float* __restrict__ cnt, int n)
{
    int i = blockIdx.x * 256 + threadIdx.x;
    if (i >= n * D1) return;
    int r = i / D1, c = i - r * D1;
    int g = batch[r];
    atomicAdd(&pooled[(size_t)g * D1 + c], x[i]);
    if (c == 0) atomicAdd(&cnt[g], 1.f);
}

// Output head: g = [pooled/cnt, params]; out = (g@W1+b1)@W2+b2 (no activation)
__global__ __launch_bounds__(128) void head_kernel(
    const float* __restrict__ pooled, const float* __restrict__ cnt,
    const float* __restrict__ pars,
    const float* __restrict__ W1, const float* __restrict__ b1,
    const float* __restrict__ W2, const float* __restrict__ b2,
    float* __restrict__ out)
{
    int g = blockIdx.x, t = threadIdx.x;
    float cc = cnt[g]; cc = cc < 1.f ? 1.f : cc;
    float hs = b1[t];
    for (int k = 0; k < D1; ++k)
        hs += (pooled[(size_t)g * D1 + k] / cc) * W1[(size_t)k * D2 + t];
    hs += pars[g * 2 + 0] * W1[(size_t)D1 * D2 + t]
        + pars[g * 2 + 1] * W1[(size_t)(D1 + 1) * D2 + t];
    __shared__ float red[128];
    red[t] = hs * W2[t];
    __syncthreads();
    for (int o = 64; o > 0; o >>= 1) {
        if (t < o) red[t] += red[t + o];
        __syncthreads();
    }
    if (t == 0) out[g] = red[0] + b2[0];
}

// ---------------------------------------------------------------------------
extern "C" void kernel_launch(void* const* d_in, const int* in_sizes, int n_in,
                              void* d_out, int out_size, void* d_ws, size_t ws_size,
                              hipStream_t stream) {
    (void)in_sizes; (void)n_in; (void)out_size; (void)ws_size;

    // Input order: dict order; nested 'params' flattened as a pytree
    // (sorted keys: att, bn, conv1, conv1_2, conv1_3, conv1_4, conv2_4,
    //  conv3_4, go1, go2, pre).
    const float* x     = (const float*)d_in[0];
    const int*   ei1   = (const int*)d_in[1];
    const int*   ei2   = (const int*)d_in[2];
    const int*   ei3   = (const int*)d_in[3];
    const float* ea1   = (const float*)d_in[4];
    const float* ea2   = (const float*)d_in[5];
    const float* ea3   = (const float*)d_in[6];
    const int*   batch = (const int*)d_in[7];
    const float* pars  = (const float*)d_in[8];
    const float* attW  = (const float*)d_in[9];
    const float* attb  = (const float*)d_in[10];
    struct Conv { const float *Wf, *bf, *Ws, *bs; };
    auto conv = [&](int base) {
        Conv c; c.Wf = (const float*)d_in[base];     c.bf = (const float*)d_in[base + 1];
                c.Ws = (const float*)d_in[base + 2]; c.bs = (const float*)d_in[base + 3];
        return c;
    };
    Conv c1  = conv(23), c12 = conv(27), c13 = conv(31);
    Conv c14 = conv(35), c24 = conv(39), c34 = conv(43);
    const float* go1W = (const float*)d_in[47];
    const float* go1b = (const float*)d_in[48];
    const float* go2W = (const float*)d_in[49];
    const float* go2b = (const float*)d_in[50];
    const float* preW[4] = {(const float*)d_in[51], (const float*)d_in[53],
                            (const float*)d_in[55], (const float*)d_in[57]};
    const float* preB[4] = {(const float*)d_in[52], (const float*)d_in[54],
                            (const float*)d_in[56], (const float*)d_in[58]};

    // ---- carve workspace ----
    char* wp = (char*)d_ws;
    auto walloc = [&](size_t bytes) {
        void* r = (void*)wp;
        wp += (bytes + 255) & ~(size_t)255;
        return r;
    };
    float* h    = (float*)walloc((size_t)NND * D1 * 4);
    float* Pfd  = (float*)walloc((size_t)NND * D1 * 4);
    float* Pfs  = (float*)walloc((size_t)NND * D1 * 4);
    float* Psd  = (float*)walloc((size_t)NND * D1 * 4);
    float* Pss  = (float*)walloc((size_t)NND * D1 * 4);
    float* agg  = (float*)walloc((size_t)NND * D1 * 4);
    float* xb   = (float*)walloc((size_t)NND * D1 * 4);
    float* xc   = (float*)walloc((size_t)NND * D1 * 4);
    float* deg  = (float*)walloc((size_t)NND * 4);
    float* A1   = (float*)walloc((size_t)NND * 4);
    float* A2   = (float*)walloc((size_t)NND * 4);
    float* eaC  = (float*)walloc((size_t)3 * NE * 4 * 4);
    int*   eiC  = (int*)  walloc((size_t)2 * 3 * NE * 4);
    float* pooled = (float*)walloc((size_t)NB * D1 * 4);
    float* cnt    = (float*)walloc((size_t)NB * 4);
    float* bsum   = (float*)walloc((size_t)D1 * 4);
    float* bsq    = (float*)walloc((size_t)D1 * 4);
    _Float16* a16 = (_Float16*)walloc((size_t)MPAD * KP * 2);
    _Float16* w16 = (_Float16*)walloc((size_t)KP * KP * 2);

    auto packA = [&](const float* A, int lda, int K) {
        pack_a_kernel<<<(MPAD * KP) / 256, 256, 0, stream>>>(A, lda, K, a16);
    };
    auto packW = [&](const float* W, int ldw, int K, int N) {
        pack_w_kernel<<<(KP * KP) / 256, 256, 0, stream>>>(W, ldw, K, N, w16);
    };
    auto gemm = [&](const float* bias, float* C, int kSteps, int act) {
        gemm_f16_wmma<<<dim3(MPAD / 64), dim3(256), 0, stream>>>(
            a16, w16, bias, C, D1, NND, D1, kSteps, act);
    };
    auto bn = [&](float* xm, int bnIdx) {
        hipMemsetAsync(bsum, 0, D1 * 4, stream);
        hipMemsetAsync(bsq, 0, D1 * 4, stream);
        bn_stats_kernel<<<(NND + ROWS_PER_BLK - 1) / ROWS_PER_BLK, 256, 0, stream>>>(
            xm, bsum, bsq, NND);
        bn_apply_kernel<<<(NND * D1 + 255) / 256, 256, 0, stream>>>(
            xm, bsum, bsq, (const float*)d_in[11 + 2 * bnIdx],
            (const float*)d_in[12 + 2 * bnIdx], NND);
    };

    // ---- pre-FC: 4 layers with ReLU ----
    packA(x, FIN, FIN); packW(preW[0], D1, FIN, D1); gemm(preB[0], h,  3, 1);
    packA(h, D1, D1);   packW(preW[1], D1, D1, D1);  gemm(preB[1], xb, 4, 1);
    packA(xb, D1, D1);  packW(preW[2], D1, D1, D1);  gemm(preB[2], xc, 4, 1);
    packA(xc, D1, D1);  packW(preW[3], D1, D1, D1);  gemm(preB[3], h,  4, 1);
    packA(h, D1, D1);   // a16 = h16, reused by all branch GEMMs + combined conv 0

    // ---- combined edge index (for convs 4-6) ----
    concat_edges<<<(3 * NE + 255) / 256, 256, 0, stream>>>(ei1, ei2, ei3, eiC);

    // ---- three parallel branches ----
    const int*   eis[3]   = {ei1, ei2, ei3};
    const float* eas[3]   = {ea1, ea2, ea3};
    Conv         convs[3] = {c1, c12, c13};
    for (int i = 0; i < 3; ++i) {
        const Conv& cv = convs[i];
        packW(cv.Wf,           D1, D1, D1); gemm(nullptr, Pfd, 4, 0);
        packW(cv.Wf + D1 * D1, D1, D1, D1); gemm(nullptr, Pfs, 4, 0);
        packW(cv.Ws,           D1, D1, D1); gemm(nullptr, Psd, 4, 0);
        packW(cv.Ws + D1 * D1, D1, D1, D1); gemm(nullptr, Pss, 4, 0);
        hipMemsetAsync(agg, 0, (size_t)NND * D1 * 4, stream);
        hipMemsetAsync(deg, 0, (size_t)NND * 4, stream);
        cgconv_edge_kernel<<<(NE + 7) / 8, 256, 0, stream>>>(
            eis[i], eis[i] + NE, eas[i], 3, Pfd, Pfs, Psd, Pss,
            cv.Wf + 2 * D1 * D1, cv.bf, cv.Ws + 2 * D1 * D1, cv.bs, agg, deg, NE);
        combine_kernel<<<(NND * D1 + 255) / 256, 256, 0, stream>>>(h, agg, deg, xb, NND);
        bn(xb, i);
        att_node_kernel<<<(NND + 255) / 256, 256, 0, stream>>>(xb, attW, A1, A2, NND);
        edge_score_kernel<<<(NE + 255) / 256, 256, 0, stream>>>(
            eis[i], eis[i] + NE, eas[i], A1, A2, attW, attb,
            eaC + (size_t)i * NE * 4, NE);
    }

    // ---- three sequential convs on combined edges (edge_dim = 4) ----
    Conv cc[3] = {c14, c24, c34};
    const float* cur = h;
    float* bufs[3] = {xb, xc, xb};
    for (int j = 0; j < 3; ++j) {
        const Conv& cv = cc[j];
        float* nxt = bufs[j];
        if (j > 0) packA(cur, D1, D1);     // j==0 uses a16 = h16 already packed
        packW(cv.Wf,           D1, D1, D1); gemm(nullptr, Pfd, 4, 0);
        packW(cv.Wf + D1 * D1, D1, D1, D1); gemm(nullptr, Pfs, 4, 0);
        packW(cv.Ws,           D1, D1, D1); gemm(nullptr, Psd, 4, 0);
        packW(cv.Ws + D1 * D1, D1, D1, D1); gemm(nullptr, Pss, 4, 0);
        hipMemsetAsync(agg, 0, (size_t)NND * D1 * 4, stream);
        hipMemsetAsync(deg, 0, (size_t)NND * 4, stream);
        cgconv_edge_kernel<<<(3 * NE + 7) / 8, 256, 0, stream>>>(
            eiC, eiC + 3 * NE, eaC, 4, Pfd, Pfs, Psd, Pss,
            cv.Wf + 2 * D1 * D1, cv.bf, cv.Ws + 2 * D1 * D1, cv.bs, agg, deg, 3 * NE);
        combine_kernel<<<(NND * D1 + 255) / 256, 256, 0, stream>>>(cur, agg, deg, nxt, NND);
        bn(nxt, 3 + j);
        cur = nxt;
    }

    // ---- global mean pool + output MLP ----
    hipMemsetAsync(pooled, 0, (size_t)NB * D1 * 4, stream);
    hipMemsetAsync(cnt, 0, (size_t)NB * 4, stream);
    pool_kernel<<<(NND * D1 + 255) / 256, 256, 0, stream>>>(cur, batch, pooled, cnt, NND);
    head_kernel<<<NB, 128, 0, stream>>>(pooled, cnt, pars, go1W, go1b, go2W, go2b,
                                        (float*)d_out);
}
